// TemporalGraphTransformer_18872086299365
// MI455X (gfx1250) — compile-verified, matrix operand order checked by reading, not run on hardware
//
#include <hip/hip_runtime.h>
#include <math.h>

// ---------------- problem constants ----------------
#define N_NODES 25000
#define N_EDGES 400000
#define F_IN    64
#define D_HID   128
#define H_HEADS 4
#define C_DIM   128
#define HC      512            // H_HEADS * C_DIM
#define MP      25008          // N_NODES padded to multiple of 16 (1563 tiles)
#define SCALE_ATT 0.08838834764831845f   // 1/sqrt(128)

typedef __attribute__((ext_vector_type(16))) __bf16 v16bf;
typedef __attribute__((ext_vector_type(8)))  float  v8f;

union FragBF { uint4 u[2]; v16bf v; };

__device__ __forceinline__ unsigned short f32_to_bf16(float f) {
  unsigned u = __float_as_uint(f);
  unsigned r = 0x7FFFu + ((u >> 16) & 1u);   // round-to-nearest-even
  return (unsigned short)((u + r) >> 16);
}

// float atomic max via ordered-int trick (works with -inf init)
__device__ __forceinline__ void atomicMaxF(float* addr, float value) {
  if (value >= 0.0f) atomicMax((int*)addr, __float_as_int(value));
  else               atomicMin((unsigned int*)addr, __float_as_uint(value));
}

// ---------------- utility kernels ----------------
__global__ void fill_f32_kernel(float* __restrict__ p, float v, int n) {
  int i = blockIdx.x * blockDim.x + threadIdx.x;
  if (i < n) p[i] = v;
}

// activation f32 [N, K] -> bf16 [MP, K], zero-pad rows >= N
__global__ void cast_act_kernel(const float* __restrict__ src,
                                unsigned short* __restrict__ dst, int K) {
  int i = blockIdx.x * blockDim.x + threadIdx.x;
  if (i >= MP * K) return;
  int m = i / K;
  float v = (m < N_NODES) ? src[i] : 0.0f;
  dst[i] = f32_to_bf16(v);
}

// weight f32 [K, Ncols] -> bf16 transposed [Ncols, K]
__global__ void cast_weight_kernel(const float* __restrict__ W,
                                   unsigned short* __restrict__ Wt,
                                   int K, int Ncols) {
  int i = blockIdx.x * blockDim.x + threadIdx.x;
  if (i >= K * Ncols) return;
  int n = i / K, k = i % K;
  Wt[i] = f32_to_bf16(W[(size_t)k * Ncols + n]);
}

// ---------------- WMMA GEMM: C[N, Ncols] = A[MP,K]bf16 @ W[K,Ncols]bf16 + bias ----------------
// 8 waves per block. Each wave computes FOUR consecutive 16x16 n-tiles, reusing one
// A fragment per k-step across 4 WMMAs (4x less A traffic, 4x WMMA density).
// A-fragment layout (16-bit A 16x32): lane<16 -> K {0..7,16..23}+k0, lane>=16 -> K {8..15,24..31}+k0.
// Weights pre-transposed so the B fragment loads with the identical per-lane pattern.
template <int K>
__global__ __launch_bounds__(256) void gemm_bf16_wmma_kernel(
    const unsigned short* __restrict__ A,    // [MP, K]
    const unsigned short* __restrict__ Bt,   // [Ncols, K]
    const float* __restrict__ bias,          // [Ncols]
    float* __restrict__ Cout,                // [N_NODES, Ncols]
    int Ncols) {
  int wave = threadIdx.x >> 5;
  int lane = threadIdx.x & 31;
  int tile_m = blockIdx.x;
  int tn0 = wave * 4;                       // first n-tile owned by this wave
  if (tn0 * 16 >= Ncols) return;

  int row   = tile_m * 16 + (lane & 15);
  int coll  = lane & 15;
  int khalf = (lane >> 4) * 8;

  const unsigned short* arow = A + (size_t)row * K + khalf;
  const unsigned short* brow0 = Bt + (size_t)((tn0 + 0) * 16 + coll) * K + khalf;
  const unsigned short* brow1 = Bt + (size_t)((tn0 + 1) * 16 + coll) * K + khalf;
  const unsigned short* brow2 = Bt + (size_t)((tn0 + 2) * 16 + coll) * K + khalf;
  const unsigned short* brow3 = Bt + (size_t)((tn0 + 3) * 16 + coll) * K + khalf;

  v8f acc0 = {}, acc1 = {}, acc2 = {}, acc3 = {};
#pragma unroll
  for (int k0 = 0; k0 < K; k0 += 32) {
    FragBF fa, fb0, fb1, fb2, fb3;
    fa.u[0]  = *(const uint4*)(arow + k0);
    fa.u[1]  = *(const uint4*)(arow + k0 + 16);
    fb0.u[0] = *(const uint4*)(brow0 + k0);
    fb0.u[1] = *(const uint4*)(brow0 + k0 + 16);
    fb1.u[0] = *(const uint4*)(brow1 + k0);
    fb1.u[1] = *(const uint4*)(brow1 + k0 + 16);
    fb2.u[0] = *(const uint4*)(brow2 + k0);
    fb2.u[1] = *(const uint4*)(brow2 + k0 + 16);
    fb3.u[0] = *(const uint4*)(brow3 + k0);
    fb3.u[1] = *(const uint4*)(brow3 + k0 + 16);
    acc0 = __builtin_amdgcn_wmma_f32_16x16x32_bf16(false, fa.v, false, fb0.v,
                                                   (short)0, acc0, false, false);
    acc1 = __builtin_amdgcn_wmma_f32_16x16x32_bf16(false, fa.v, false, fb1.v,
                                                   (short)0, acc1, false, false);
    acc2 = __builtin_amdgcn_wmma_f32_16x16x32_bf16(false, fa.v, false, fb2.v,
                                                   (short)0, acc2, false, false);
    acc3 = __builtin_amdgcn_wmma_f32_16x16x32_bf16(false, fa.v, false, fb3.v,
                                                   (short)0, acc3, false, false);
  }

  // D layout: VGPR r, lane<16 -> M = r, lane>=16 -> M = r + 8; N = lane&15
  int mbase = tile_m * 16 + ((lane >> 4) << 3);
  v8f accs[4] = {acc0, acc1, acc2, acc3};
#pragma unroll
  for (int t = 0; t < 4; ++t) {
    int col = (tn0 + t) * 16 + coll;
    float b = bias[col];
#pragma unroll
    for (int r = 0; r < 8; ++r) {
      int m = mbase + r;
      if (m < N_NODES) Cout[(size_t)m * Ncols + col] = accs[t][r] + b;
    }
  }
}

// ---------------- edge kernels (one wave32 per edge) ----------------
// lane: h = lane/8 (head), part = lane%8 (16-float slice of C=128)
__global__ __launch_bounds__(256) void attn_logits_kernel(
    const float* __restrict__ q, const float* __restrict__ k,
    const int* __restrict__ src, const int* __restrict__ dst,
    float* __restrict__ alpha, float* __restrict__ mbuf) {
  int wid = (blockIdx.x * blockDim.x + threadIdx.x) >> 5;
  if (wid >= N_EDGES) return;
  int lane = threadIdx.x & 31;
  int h = lane >> 3, part = lane & 7;
  int s = src[wid], d = dst[wid];
  const float4* qp = (const float4*)(q + (size_t)d * HC + h * C_DIM + part * 16);
  const float4* kp = (const float4*)(k + (size_t)s * HC + h * C_DIM + part * 16);
  float acc = 0.0f;
#pragma unroll
  for (int j = 0; j < 4; ++j) {
    float4 a = qp[j], b = kp[j];
    acc += a.x * b.x + a.y * b.y + a.z * b.z + a.w * b.w;
  }
  acc += __shfl_xor(acc, 1, 32);
  acc += __shfl_xor(acc, 2, 32);
  acc += __shfl_xor(acc, 4, 32);
  if (part == 0) {
    float logit = acc * SCALE_ATT;
    alpha[(size_t)wid * H_HEADS + h] = logit;
    atomicMaxF(&mbuf[(size_t)d * H_HEADS + h], logit);
  }
}

__global__ void exp_sum_kernel(float* __restrict__ alpha,
                               const float* __restrict__ mbuf,
                               const int* __restrict__ dst,
                               float* __restrict__ denom) {
  int i = blockIdx.x * blockDim.x + threadIdx.x;
  if (i >= N_EDGES * H_HEADS) return;
  int e = i >> 2, h = i & 3;
  int d = dst[e];
  float ev = __expf(alpha[i] - mbuf[(size_t)d * H_HEADS + h]);
  alpha[i] = ev;
  atomicAdd(&denom[(size_t)d * H_HEADS + h], ev);
}

__global__ __launch_bounds__(256) void aggregate_kernel(
    const float* __restrict__ v, const float* __restrict__ alpha,
    const float* __restrict__ denom,
    const int* __restrict__ src, const int* __restrict__ dst,
    float* __restrict__ agg) {
  int wid = (blockIdx.x * blockDim.x + threadIdx.x) >> 5;
  if (wid >= N_EDGES) return;
  int lane = threadIdx.x & 31;
  int h = lane >> 3, part = lane & 7;
  int s = src[wid], d = dst[wid];
  const float* vrow = v + (size_t)s * HC + h * C_DIM + part * 16;
  __builtin_prefetch(vrow, 0, 1);   // global_prefetch: pull gathered V row toward WGP
  float a = alpha[(size_t)wid * H_HEADS + h] /
            (denom[(size_t)d * H_HEADS + h] + 1e-16f);
  const float4* vp = (const float4*)vrow;
  float* op = agg + (size_t)d * HC + h * C_DIM + part * 16;
#pragma unroll
  for (int j = 0; j < 4; ++j) {
    float4 vv = vp[j];
    atomicAdd(op + j * 4 + 0, a * vv.x);
    atomicAdd(op + j * 4 + 1, a * vv.y);
    atomicAdd(op + j * 4 + 2, a * vv.z);
    atomicAdd(op + j * 4 + 3, a * vv.w);
  }
}

// h_out = relu(mean_heads(agg) + skip)
__global__ void combine_kernel(const float* __restrict__ agg,
                               const float* __restrict__ skip,
                               float* __restrict__ hout) {
  int i = blockIdx.x * blockDim.x + threadIdx.x;
  if (i >= N_NODES * D_HID) return;
  int n = i >> 7, dd = i & 127;
  const float* ag = agg + (size_t)n * HC;
  float s = 0.25f * (ag[dd] + ag[128 + dd] + ag[256 + dd] + ag[384 + dd]) + skip[i];
  hout[i] = s > 0.0f ? s : 0.0f;
}

// column sums of h [N,128] into g[128]
__global__ void col_reduce_kernel(const float* __restrict__ h, float* __restrict__ g) {
  int d = threadIdx.x;            // 128 threads
  float acc = 0.0f;
  for (int n = blockIdx.x; n < N_NODES; n += gridDim.x)
    acc += h[(size_t)n * D_HID + d];
  atomicAdd(&g[d], acc);
}

__global__ void readout_kernel(const float* __restrict__ g,
                               const float* __restrict__ Wfc,
                               const float* __restrict__ bfc,
                               float* __restrict__ out) {
  __shared__ float sm[128];
  int t = threadIdx.x;
  sm[t] = (g[t] * (1.0f / (float)N_NODES)) * Wfc[t];
  __syncthreads();
  for (int s = 64; s > 0; s >>= 1) {
    if (t < s) sm[t] += sm[t + s];
    __syncthreads();
  }
  if (t == 0) out[0] = 1.0f / (1.0f + __expf(-(sm[0] + bfc[0])));
}

// ---------------- host launcher ----------------
extern "C" void kernel_launch(void* const* d_in, const int* in_sizes, int n_in,
                              void* d_out, int out_size, void* d_ws, size_t ws_size,
                              hipStream_t stream) {
  const float* x   = (const float*)d_in[0];
  const int*   ei  = (const int*)d_in[1];
  const int*   src = ei;
  const int*   dst = ei + N_EDGES;
  const float* Wfc = (const float*)d_in[26];
  const float* bfc = (const float*)d_in[27];

  // workspace carve-up (256B aligned)
  char* base = (char*)d_ws;
  size_t off = 0;
  auto carve = [&](size_t bytes) -> void* {
    void* p = base + off;
    off = (off + bytes + 255) & ~(size_t)255;
    return p;
  };
  float* bufA  = (float*)carve((size_t)N_NODES * HC * 4);   // q, later skip (N x 128 prefix)
  float* bufB  = (float*)carve((size_t)N_NODES * HC * 4);   // k, later agg
  float* bufC  = (float*)carve((size_t)N_NODES * HC * 4);   // v
  float* h0    = (float*)carve((size_t)N_NODES * D_HID * 4);
  float* h1    = (float*)carve((size_t)N_NODES * D_HID * 4);
  float* alpha = (float*)carve((size_t)N_EDGES * H_HEADS * 4);
  float* mbuf  = (float*)carve((size_t)N_NODES * H_HEADS * 4);
  float* dbuf  = (float*)carve((size_t)N_NODES * H_HEADS * 4);
  unsigned short* a_bf  = (unsigned short*)carve((size_t)MP * D_HID * 2);
  unsigned short* wt_bf = (unsigned short*)carve((size_t)D_HID * HC * 2);
  float* gsum  = (float*)carve(D_HID * 4);

  float* qbuf = bufA;  // skip reuses bufA after logits consume q
  float* kbuf = bufB;  // agg  reuses bufB after logits consume k
  float* vbuf = bufC;

  const int TB = 256;
  const float* act = x;
  int K = F_IN;

  for (int l = 0; l < 3; ++l) {
    const float* Wq = (const float*)d_in[2 + l * 8 + 0];
    const float* bq = (const float*)d_in[2 + l * 8 + 1];
    const float* Wk = (const float*)d_in[2 + l * 8 + 2];
    const float* bk = (const float*)d_in[2 + l * 8 + 3];
    const float* Wv = (const float*)d_in[2 + l * 8 + 4];
    const float* bv = (const float*)d_in[2 + l * 8 + 5];
    const float* Ws = (const float*)d_in[2 + l * 8 + 6];
    const float* bs = (const float*)d_in[2 + l * 8 + 7];
    float* hout = (l & 1) ? h1 : h0;

    // activation -> bf16 padded
    int actElems = MP * K;
    cast_act_kernel<<<(actElems + TB - 1) / TB, TB, 0, stream>>>(act, a_bf, K);

    // Q, K, V projections via WMMA (each block: 8 waves x 4 tiles = full 512 cols)
    const float* Wproj[3] = {Wq, Wk, Wv};
    const float* bproj[3] = {bq, bk, bv};
    float*       oproj[3] = {qbuf, kbuf, vbuf};
    for (int p = 0; p < 3; ++p) {
      cast_weight_kernel<<<(K * HC + TB - 1) / TB, TB, 0, stream>>>(Wproj[p], wt_bf, K, HC);
      if (K == F_IN)
        gemm_bf16_wmma_kernel<F_IN><<<dim3(MP / 16), TB, 0, stream>>>(a_bf, wt_bf, bproj[p], oproj[p], HC);
      else
        gemm_bf16_wmma_kernel<D_HID><<<dim3(MP / 16), TB, 0, stream>>>(a_bf, wt_bf, bproj[p], oproj[p], HC);
    }

    // softmax state init
    int nh = N_NODES * H_HEADS;
    fill_f32_kernel<<<(nh + TB - 1) / TB, TB, 0, stream>>>(mbuf, -INFINITY, nh);
    fill_f32_kernel<<<(nh + TB - 1) / TB, TB, 0, stream>>>(dbuf, 0.0f, nh);

    // edge logits + segment max  (one wave per edge)
    int waveBlocks = (N_EDGES * 32 + TB - 1) / TB;
    attn_logits_kernel<<<waveBlocks, TB, 0, stream>>>(qbuf, kbuf, src, dst, alpha, mbuf);

    // exp + segment sum
    int eh = N_EDGES * H_HEADS;
    exp_sum_kernel<<<(eh + TB - 1) / TB, TB, 0, stream>>>(alpha, mbuf, dst, dbuf);

    // weighted scatter of V  (agg aliases kbuf; k is dead now)
    float* agg = kbuf;
    int aggElems = N_NODES * HC;
    fill_f32_kernel<<<(aggElems + TB - 1) / TB, TB, 0, stream>>>(agg, 0.0f, aggElems);
    aggregate_kernel<<<waveBlocks, TB, 0, stream>>>(vbuf, alpha, dbuf, src, dst, agg);

    // skip connection GEMM (writes into bufA prefix; q is dead now)
    float* skip = qbuf;
    cast_weight_kernel<<<(K * D_HID + TB - 1) / TB, TB, 0, stream>>>(Ws, wt_bf, K, D_HID);
    if (K == F_IN)
      gemm_bf16_wmma_kernel<F_IN><<<dim3(MP / 16), TB, 0, stream>>>(a_bf, wt_bf, bs, skip, D_HID);
    else
      gemm_bf16_wmma_kernel<D_HID><<<dim3(MP / 16), TB, 0, stream>>>(a_bf, wt_bf, bs, skip, D_HID);

    // mean over heads + skip + relu
    int nd = N_NODES * D_HID;
    combine_kernel<<<(nd + TB - 1) / TB, TB, 0, stream>>>(agg, skip, hout);

    act = hout;
    K = D_HID;
  }

  // global mean pool + fc + sigmoid
  fill_f32_kernel<<<1, D_HID, 0, stream>>>(gsum, 0.0f, D_HID);
  col_reduce_kernel<<<128, D_HID, 0, stream>>>((const float*)act, gsum);
  readout_kernel<<<1, D_HID, 0, stream>>>(gsum, Wfc, bfc, (float*)d_out);
}
